// ContrastiveAttention_26036091748420
// MI455X (gfx1250) — matrix operand order repaired and checked by reading, not compiled, not run on hardware
//
#include <hip/hip_runtime.h>

// ---------------------------------------------------------------------------
// Fused contrastive (softmin) attention for MI455X / gfx1250, wave32.
//   scores[b,q,d] = sum_m Q[q,m] * M[b,d,m]
//   attn = softmax(-scores, axis=d)   (softmin)
//   out[b,q,m]   = sum_d attn[b,q,d] * M[b,d,m]
// Flash-style single pass over M, bf16 WMMA (f32 accumulate), with the
// Tensor Data Mover (TDM) streaming the next M chunk into LDS while the
// current chunk is consumed by v_wmma_f32_16x16x32_bf16.
// ---------------------------------------------------------------------------

typedef __attribute__((ext_vector_type(16))) __bf16    v16bf;
typedef __attribute__((ext_vector_type(8)))  __bf16    v8bf;
typedef __attribute__((ext_vector_type(8)))  float     v8f;
typedef __attribute__((ext_vector_type(4)))  unsigned  v4u;
typedef __attribute__((ext_vector_type(8)))  int       v8i;
typedef __attribute__((ext_vector_type(4)))  int       v4i;

#define Bsz   128   // batches
#define Dsz   1024  // memory slots
#define DMsz  1024  // model dim
#define Qsz   128   // query rows
#define QT    32    // query rows per workgroup
#define DC    32    // memory slots per chunk
#define NCH   (Dsz / DC)
#define QS_STR 40   // qsT / pt row stride (halfs): 80B, 16B aligned, bank-friendly
#define MS_STR 1032 // msb row stride (halfs): 2064B, 16B aligned, bank-friendly

// float -> bf16 with round-to-nearest-even (packed pair -> u32)
__device__ __forceinline__ unsigned pk2bf(float a, float b) {
  unsigned ua = __builtin_bit_cast(unsigned, a);
  unsigned ub = __builtin_bit_cast(unsigned, b);
  ua = (ua + 0x7FFFu + ((ua >> 16) & 1u)) >> 16;
  ub = (ub + 0x7FFFu + ((ub >> 16) & 1u)) >> 16;
  return (ua & 0xFFFFu) | (ub << 16);
}
__device__ __forceinline__ __bf16 cvtbf(float a) {
  unsigned ua = __builtin_bit_cast(unsigned, a);
  ua = (ua + 0x7FFFu + ((ua >> 16) & 1u)) >> 16;
  unsigned short s = (unsigned short)ua;
  return __builtin_bit_cast(__bf16, s);
}

// Two contiguous 16B LDS loads -> one 16-element bf16 fragment.
// off2 = 16 for A fragments (K runs 0..7 then 16..23); B fragments are gathered in the same k order.
__device__ __forceinline__ v16bf ldpair(const __bf16* p, int off2) {
  v8bf lo = *(const v8bf*)p;
  v8bf hi = *(const v8bf*)(p + off2);
  return __builtin_shufflevector(lo, hi, 0, 1, 2, 3, 4, 5, 6, 7,
                                         8, 9, 10, 11, 12, 13, 14, 15);
}

__device__ __forceinline__ v8f wmma_bf16(v16bf a, v16bf b, v8f c) {
  return __builtin_amdgcn_wmma_f32_16x16x32_bf16(
      /*neg_a=*/false, a, /*neg_b=*/false, b,
      /*c_mod=*/(short)0, c, /*reuse_a=*/false, /*reuse_b=*/false);
}

// Issue one TDM 2D tile load: DC x 1024 f32, global row-major -> LDS.
// D# packing per CDNA5 ISA ch.8 (group0: count/lds/global/type, group1: dims).
// This toolchain exposes the 6-arg builtin:
//   (v4u g0, v8i g1, v4i g2, v4i g3, v8i g4, i32 cpol)
__device__ __forceinline__ void tdm_load_chunk(const float* gaddr, unsigned lds_off) {
  unsigned long long ga = (unsigned long long)(uintptr_t)gaddr;
  v4u g0;
  g0[0] = 1u;                                   // count=1, user descriptor
  g0[1] = lds_off;                              // lds_addr (bytes)
  g0[2] = (unsigned)(ga & 0xFFFFFFFFu);         // global_addr[31:0]
  g0[3] = (unsigned)((ga >> 32) & 0x1FFFFFFu)   // global_addr[56:32]
          | 0x80000000u;                        // type = 2 ("image")
  v8i g1;
  g1[0] = 0x20000;                              // data_size = 2 (4 bytes)
  g1[1] = (int)(1024u << 16);                   // tensor_dim0 = 1024 (lo16)
  g1[2] = (int)(1024u << 16);                   // dim0 hi=0 | tensor_dim1 = 1024 (lo16)
  g1[3] = (int)(1024u << 16);                   // dim1 hi=0 | tile_dim0 = 1024
  g1[4] = DC;                                   // tile_dim1 = 32, tile_dim2 = 0
  g1[5] = 1024;                                 // tensor_dim0_stride = 1024
  g1[6] = 0;                                    // stride hi | dim1_stride lo
  g1[7] = 0;
  v4i z4 = {};
  v8i z8 = {};
  __builtin_amdgcn_tensor_load_to_lds(g0, g1, z4, z4, z8, 0);
}

__global__ __launch_bounds__(256, 1)
void contrastive_attn_kernel(const float* __restrict__ query,
                             const float* __restrict__ memory,
                             float* __restrict__ out) {
  // LDS: 131072 + 81920 + 66048 + 8448 + 2560 + 384 = 290432 B (<320KB/WGP)
  __shared__ __align__(16) float  stg[DC][DMsz];       // TDM staging (f32 chunk)
  __shared__ __align__(16) __bf16 qsT[DMsz][QS_STR];   // Q tile, transposed (k-major)
  __shared__ __align__(16) __bf16 msb[DC][MS_STR];     // M chunk, row-major bf16
  __shared__ __align__(16) float  sred[2][DC][33];     // S^T k-half partials / reduced
  __shared__ __align__(16) __bf16 pt[QT][QS_STR];      // P = exp(m - s), row-major
  __shared__ float mrow[QT], lrow[QT], srow[QT];       // running min / sum / rescale

  const int tid  = threadIdx.x;
  const int lane = tid & 31;
  const int wv   = tid >> 5;                 // 8 waves
  const int b    = blockIdx.x >> 2;
  const int q0   = (blockIdx.x & 3) * QT;

  const float* memG = memory + (size_t)b * Dsz * DMsz;
  const unsigned stg_base = (unsigned)(uintptr_t)(void*)&stg[0][0];

  // Kick off the DMA of chunk 0 before touching Q (overlaps prologue).
  if (wv == 0) tdm_load_chunk(memG, stg_base);

  // ---- Prologue: Q tile -> qsT (bf16, transposed) ----
  for (int i = tid; i < QT * DMsz; i += 256) {
    int r = i >> 10, k = i & (DMsz - 1);
    qsT[k][r] = cvtbf(query[(q0 + r) * DMsz + k]);
  }
  if (tid < QT) { mrow[tid] = __builtin_inff(); lrow[tid] = 0.f; }

  // Per-wave WMMA geometry
  const int kb    = (lane >> 4) << 3;        // A-frag K sub-offset: 0 or 8
  const int tile  = wv >> 1;                 // S^T tile owned by wave pair
  const int khalf = wv & 1;                  // K range half (0..511 / 512..1023)
  const int smi   = tile & 1;                // S^T M-tile (slot rows)
  const int sni   = tile >> 1;               // S^T N-tile (query rows)
  const int arow  = smi * 16 + (lane & 15);  // A row (slot) for S gemm
  const int crow  = smi * 16 + ((lane >> 4) << 3);  // C row base
  const int ccol  = sni * 16 + (lane & 15);         // C col

  v8f zero = {};
  v8f oacc[2][8];                            // 32x128 f32 output accum per wave
  #pragma unroll
  for (int mi = 0; mi < 2; ++mi)
    #pragma unroll
    for (int nj = 0; nj < 8; ++nj) oacc[mi][nj] = zero;

  for (int ch = 0; ch < NCH; ++ch) {
    const int d0 = ch * DC;

    // ---- Wait for DMA of chunk ch, then convert stg(f32) -> msb(bf16) ----
    if (wv == 0) __builtin_amdgcn_s_wait_tensorcnt(0);
    __syncthreads();

    #pragma unroll 4
    for (int j = 0; j < 32; ++j) {
      int i4   = tid + j * 256;
      int slot = i4 >> 8;
      int col  = (i4 & 255) << 2;
      const float4 f = *(const float4*)&stg[slot][col];
      unsigned* dst = (unsigned*)&msb[slot][col];
      dst[0] = pk2bf(f.x, f.y);
      dst[1] = pk2bf(f.z, f.w);
    }
    __syncthreads();

    // ---- Issue DMA of chunk ch+1; overlaps all compute below ----
    if (ch + 1 < NCH && wv == 0)
      tdm_load_chunk(memG + (size_t)(d0 + DC) * DMsz, stg_base);

    // ---- S^T = M_chunk (32x1024) x Q^T (1024x32), k-split per wave pair ----
    v8f sacc = zero;
    const int k0 = khalf * 512;
    #pragma unroll 4
    for (int k = k0; k < k0 + 512; k += 32) {
      v16bf a  = ldpair(&msb[arow][k + kb], 16);        // A: M chunk rows
      v16bf bq;                                         // B: Q^T (k-major)
      #pragma unroll
      for (int s = 0; s < 16; ++s)
        bq[s] = qsT[k + kb + (s & 7) + ((s >> 3) << 4)][sni * 16 + (lane & 15)];
      sacc = wmma_bf16(a, bq, sacc);
    }
    #pragma unroll
    for (int v = 0; v < 8; ++v) sred[khalf][crow + v][ccol] = sacc[v];
    __syncthreads();

    // ---- Reduce the two K halves ----
    {
      int e0 = tid << 2;
      #pragma unroll
      for (int e = e0; e < e0 + 4; ++e) {
        int r = e >> 5, c = e & 31;
        sred[0][r][c] += sred[1][r][c];
      }
    }
    __syncthreads();

    // ---- Online softmin update (one thread per query row) ----
    if (tid < QT) {
      float mo = mrow[tid];
      float mn = mo;
      #pragma unroll 8
      for (int d = 0; d < DC; ++d) mn = fminf(mn, sred[0][d][tid]);
      float sc  = __expf(mn - mo);       // first chunk: exp(-inf) = 0
      float sum = 0.f;
      #pragma unroll 8
      for (int d = 0; d < DC; ++d) {
        float p = __expf(mn - sred[0][d][tid]);
        sum += p;
        pt[tid][d] = cvtbf(p);
      }
      lrow[tid] = lrow[tid] * sc + sum;
      mrow[tid] = mn;
      srow[tid] = sc;
    }
    __syncthreads();

    // ---- Rescale accumulators, then O += P (32x32) x M_chunk (32x1024) ----
    float scv[2][8];
    #pragma unroll
    for (int mi = 0; mi < 2; ++mi)
      #pragma unroll
      for (int v = 0; v < 8; ++v)
        scv[mi][v] = srow[mi * 16 + ((lane >> 4) << 3) + v];
    #pragma unroll
    for (int mi = 0; mi < 2; ++mi)
      #pragma unroll
      for (int nj = 0; nj < 8; ++nj)
        #pragma unroll
        for (int v = 0; v < 8; ++v)
          oacc[mi][nj][v] *= scv[mi][v];

    v16bf pa0 = ldpair(&pt[(lane & 15)][kb], 16);
    v16bf pa1 = ldpair(&pt[16 + (lane & 15)][kb], 16);
    #pragma unroll
    for (int nj = 0; nj < 8; ++nj) {
      v16bf bm;                                              // B: M chunk, n=lane&15
      #pragma unroll
      for (int s = 0; s < 16; ++s)
        bm[s] = msb[kb + (s & 7) + ((s >> 3) << 4)][wv * 128 + nj * 16 + (lane & 15)];
      oacc[0][nj] = wmma_bf16(pa0, bm, oacc[0][nj]);
      oacc[1][nj] = wmma_bf16(pa1, bm, oacc[1][nj]);
    }
    __syncthreads();   // protect msb/pt/sred/stg before next chunk
  }

  // ---- Epilogue: out = O / l  (nontemporal: written once, keep L2 for M) ----
  float invl[2][8];
  #pragma unroll
  for (int mi = 0; mi < 2; ++mi)
    #pragma unroll
    for (int v = 0; v < 8; ++v)
      invl[mi][v] = 1.0f / lrow[mi * 16 + ((lane >> 4) << 3) + v];

  #pragma unroll
  for (int mi = 0; mi < 2; ++mi)
    #pragma unroll
    for (int nj = 0; nj < 8; ++nj)
      #pragma unroll
      for (int v = 0; v < 8; ++v) {
        int row = q0 + mi * 16 + ((lane >> 4) << 3) + v;
        int col = wv * 128 + nj * 16 + (lane & 15);
        __builtin_nontemporal_store(oacc[mi][nj][v] * invl[mi][v],
                                    &out[((size_t)b * Qsz + row) * DMsz + col]);
      }
}

extern "C" void kernel_launch(void* const* d_in, const int* in_sizes, int n_in,
                              void* d_out, int out_size, void* d_ws, size_t ws_size,
                              hipStream_t stream) {
  const float* query  = (const float*)d_in[0];   // [128,1024] f32
  const float* memory = (const float*)d_in[1];   // [128,1024,1024] f32
  float* out = (float*)d_out;                    // [128,128,1024] f32
  (void)in_sizes; (void)n_in; (void)out_size; (void)d_ws; (void)ws_size;

  dim3 grid(Bsz * (Qsz / QT));   // 512 workgroups
  dim3 block(256);               // 8 wave32 waves
  contrastive_attn_kernel<<<grid, block, 0, stream>>>(query, memory, out);
}
